// LSTM_Classifier_45183055954323
// MI455X (gfx1250) — matrix operand order repaired and checked by reading, not compiled, not run on hardware
//
#include <hip/hip_runtime.h>
#include <hip/hip_bf16.h>

// ---------------------------------------------------------------------------
// CDNA5 (gfx1250, wave32) fused LSTM + classifier.
//   Layer kernels: one wave (32 lanes) owns a 16-row batch tile and carries
//   h,c across all T steps. Gate pre-activations computed with
//   v_wmma_f32_16x16x32_f16: A = [16 x K] activations, B = [K x 16] weight
//   tiles (5 N-tiles cover 4H = 80), C initialized with the bias.
// ---------------------------------------------------------------------------

typedef __attribute__((ext_vector_type(16))) _Float16 v16h;
typedef __attribute__((ext_vector_type(8)))  float    v8f;

// ---- WMMA fragment loaders (CDNA5 16-bit layouts, CDNA5 ISA) ----
// A (16x32 f16): lane l -> M = l&15, half = l>>4; element j -> K = 16*(j>>3) + (j&7) + 8*half
template <typename ST>
__device__ inline v16h load_a_frag(const ST* __restrict__ src, size_t ld, int kmax) {
  const int lane = (int)(threadIdx.x & 31);
  const int m    = lane & 15;
  const int hf   = lane >> 4;
  v16h a;
#pragma unroll
  for (int j = 0; j < 16; ++j) {
    const int k = ((j >> 3) << 4) + (j & 7) + (hf << 3);
    a[j] = (k < kmax) ? (_Float16)src[(size_t)m * ld + (size_t)k] : (_Float16)0.0f;
  }
  return a;
}

// B (32x16 f16, src is W[n,k] row-major, 16 rows starting at src):
// lane l -> N = l&15, half = l>>4; element j -> K = 16*(j>>3) + (j&7) + 8*half
__device__ inline v16h load_b_frag(const float* __restrict__ src, size_t ld, int kmax) {
  const int lane = (int)(threadIdx.x & 31);
  const int n    = lane & 15;
  const int hf   = lane >> 4;
  v16h b;
#pragma unroll
  for (int j = 0; j < 16; ++j) {
    const int k = ((j >> 3) << 4) + (j & 7) + (hf << 3);
    b[j] = (k < kmax) ? (_Float16)src[(size_t)n * ld + (size_t)k] : (_Float16)0.0f;
  }
  return b;
}

__device__ inline float sigmoid_f(float x) { return 1.0f / (1.0f + __expf(-x)); }

// ---------------------------------------------------------------------------
// Fused LSTM layer.
//   FIN       : input feature width (64 for layer 0, 20 for layer 1)
//   IN_HALF   : input is f16 time-major [T][B][FIN]; else f32 [B][T][FIN]
//   OUT_HALF  : output h stored f16 time-major [T][B][20] (feeds next layer),
//               else f32 batch-major [B][T*20] (feeds the classifier)
// Grid: B/16 blocks of 32 threads (one full wave each; EXEC all ones for WMMA).
// ---------------------------------------------------------------------------
template <int FIN, bool IN_HALF, bool OUT_HALF>
__global__ __launch_bounds__(32)
void lstm_wmma_kernel(const void* __restrict__ in_seq,
                      const float* __restrict__ W_ih,   // [80, FIN]
                      const float* __restrict__ W_hh,   // [80, 20]
                      const float* __restrict__ bias,   // [80]
                      void* __restrict__ out_seq,
                      int B, int T) {
  constexpr int KCH = (FIN + 31) / 32;       // 32-wide K chunks for input GEMM
  __shared__ float    pre[16][80];           // gate pre-activations (D tiles)
  __shared__ float    cst[16][20];           // cell state
  __shared__ _Float16 hst[16][32];           // h state, f16, K-padded to 32

  const int lane = (int)threadIdx.x;
  const int b0   = (int)blockIdx.x * 16;

  // zero-init state (hst[.][20..31] stays zero forever -> pads K for WMMA)
  for (int i = lane; i < 16 * 20; i += 32) (&cst[0][0])[i] = 0.0f;
  for (int i = lane; i < 16 * 32; i += 32) (&hst[0][0])[i] = (_Float16)0.0f;
  __syncthreads();

  // Register-resident weight fragments for the whole time loop.
  v16h bih[5][KCH];
  v16h bhh[5];
#pragma unroll
  for (int nt = 0; nt < 5; ++nt) {
#pragma unroll
    for (int kc = 0; kc < KCH; ++kc)
      bih[nt][kc] = load_b_frag(W_ih + (size_t)(nt * 16) * FIN + kc * 32,
                                (size_t)FIN, FIN - kc * 32);
    bhh[nt] = load_b_frag(W_hh + (size_t)(nt * 16) * 20, (size_t)20, 20);
  }
  float bv[5];
#pragma unroll
  for (int nt = 0; nt < 5; ++nt) bv[nt] = bias[nt * 16 + (lane & 15)];

  for (int t = 0; t < T; ++t) {
    // ---- input A fragments ----
    v16h af[KCH];
    if (IN_HALF) {
      const _Float16* src = (const _Float16*)in_seq + ((size_t)t * B + b0) * FIN;
#pragma unroll
      for (int kc = 0; kc < KCH; ++kc)
        af[kc] = load_a_frag<_Float16>(src + kc * 32, (size_t)FIN, FIN - kc * 32);
      if (t + 1 < T)  // global_prefetch_b8 for next step's slab
        __builtin_prefetch(
            (const void*)((const _Float16*)in_seq +
                          ((size_t)(t + 1) * B + b0 + (lane & 15)) * FIN), 0, 1);
    } else {
      const float* src = (const float*)in_seq + ((size_t)b0 * T + t) * FIN;
#pragma unroll
      for (int kc = 0; kc < KCH; ++kc)
        af[kc] = load_a_frag<float>(src + kc * 32, (size_t)T * FIN, FIN - kc * 32);
      if (t + 1 < T)
        __builtin_prefetch(
            (const void*)((const float*)in_seq +
                          ((size_t)(b0 + (lane & 15)) * T + t + 1) * FIN), 0, 1);
    }

    // h(t-1) fragment from the LDS staging tile (zeros pad K 20..31)
    v16h hfrag = load_a_frag<_Float16>(&hst[0][0], (size_t)32, 32);

    // ---- 5 N-tiles of the fused gate GEMM: pre = x@Wih^T + h@Whh^T + b ----
#pragma unroll
    for (int nt = 0; nt < 5; ++nt) {
      v8f c;
#pragma unroll
      for (int r = 0; r < 8; ++r) c[r] = bv[nt];
#pragma unroll
      for (int kc = 0; kc < KCH; ++kc)
        c = __builtin_amdgcn_wmma_f32_16x16x32_f16(false, af[kc],
                                                   false, bih[nt][kc],
                                                   (short)0, c, false, false);
      c = __builtin_amdgcn_wmma_f32_16x16x32_f16(false, hfrag,
                                                 false, bhh[nt],
                                                 (short)0, c, false, false);
      // D layout: VGPR r, lane l -> M = r + 8*(l>>4), N = l&15
#pragma unroll
      for (int r = 0; r < 8; ++r)
        pre[r + ((lane >> 4) << 3)][nt * 16 + (lane & 15)] = c[r];
    }
    __syncthreads();

    // ---- gates: 16 rows x 20 units = 320, 10 per lane ----
#pragma unroll
    for (int u = 0; u < 10; ++u) {
      const int idx = lane * 10 + u;
      const int m   = idx / 20;
      const int j   = idx - m * 20;
      const float gi = pre[m][j];
      const float gf = pre[m][20 + j];
      const float gg = pre[m][40 + j];
      const float go = pre[m][60 + j];
      const float cn = sigmoid_f(gf) * cst[m][j] + sigmoid_f(gi) * tanhf(gg);
      const float hn = sigmoid_f(go) * tanhf(cn);
      cst[m][j] = cn;
      hst[m][j] = (_Float16)hn;
      if (OUT_HALF) {
        ((_Float16*)out_seq)[((size_t)t * B + b0 + m) * 20 + j] = (_Float16)hn;
      } else {
        ((float*)out_seq)[(size_t)(b0 + m) * ((size_t)T * 20) +
                          (size_t)t * 20 + j] = hn;
      }
    }
    __syncthreads();
  }
}

// ---------------------------------------------------------------------------
// Classifier: out[b,c] = b_fc[c] + sum_k hs1[b,k] * W_fc[c,k], k = T*20
// ---------------------------------------------------------------------------
__global__ __launch_bounds__(256)
void fc_kernel(const float* __restrict__ hs1,   // [B, T*20]
               const float* __restrict__ W_fc,  // [10, T*20]
               const float* __restrict__ b_fc,  // [10]
               float* __restrict__ out,         // [B, 10]
               int B, int T) {
  constexpr int C = 10;
  __shared__ float red[C][256];
  const int b   = (int)blockIdx.x;
  const int tid = (int)threadIdx.x;
  const int KT  = T * 20;

  float acc[C];
#pragma unroll
  for (int c = 0; c < C; ++c) acc[c] = 0.0f;

  for (int k = tid; k < KT; k += 256) {
    const float hv = hs1[(size_t)b * KT + k];
#pragma unroll
    for (int c = 0; c < C; ++c) acc[c] += hv * W_fc[(size_t)c * KT + k];
  }
#pragma unroll
  for (int c = 0; c < C; ++c) red[c][tid] = acc[c];
  __syncthreads();
  for (int s = 128; s > 0; s >>= 1) {
    if (tid < s) {
#pragma unroll
      for (int c = 0; c < C; ++c) red[c][tid] += red[c][tid + s];
    }
    __syncthreads();
  }
  if (tid < C) out[(size_t)b * C + tid] = red[tid][0] + b_fc[tid];
}

// ---------------------------------------------------------------------------
extern "C" void kernel_launch(void* const* d_in, const int* in_sizes, int n_in,
                              void* d_out, int out_size, void* d_ws, size_t ws_size,
                              hipStream_t stream) {
  const float* x     = (const float*)d_in[0];  // [B, T, 64]
  const float* W_ih0 = (const float*)d_in[1];  // [80, 64]
  const float* W_hh0 = (const float*)d_in[2];  // [80, 20]
  const float* b0    = (const float*)d_in[3];  // [80]
  const float* W_ih1 = (const float*)d_in[4];  // [80, 20]
  const float* W_hh1 = (const float*)d_in[5];  // [80, 20]
  const float* b1    = (const float*)d_in[6];  // [80]
  const float* W_fc  = (const float*)d_in[7];  // [10, T*20]
  const float* b_fc  = (const float*)d_in[8];  // [10]
  float* out = (float*)d_out;

  const int C = in_sizes[8];               // 10
  const int F = in_sizes[1] / 80;          // 64
  const int T = in_sizes[7] / (C * 20);    // 512
  const int B = in_sizes[0] / (T * F);     // 1024

  // Workspace layout: hs0 (f16, [T][B][20]) then hs1 (f32, [B][T*20])
  _Float16* hs0 = (_Float16*)d_ws;
  size_t hs0_bytes = ((size_t)T * B * 20 * sizeof(_Float16) + 255) & ~(size_t)255;
  float* hs1 = (float*)((char*)d_ws + hs0_bytes);

  dim3 wave(32);
  // Layer 0: f32 [B,T,64] input -> f16 [T,B,20] hidden states
  lstm_wmma_kernel<64, false, true><<<B / 16, wave, 0, stream>>>(
      (const void*)x, W_ih0, W_hh0, b0, (void*)hs0, B, T);
  // Layer 1: f16 [T,B,20] input -> f32 [B,T*20] hidden states
  lstm_wmma_kernel<20, true, false><<<B / 16, wave, 0, stream>>>(
      (const void*)hs0, W_ih1, W_hh1, b1, (void*)hs1, B, T);
  // Classifier
  fc_kernel<<<B, 256, 0, stream>>>(hs1, W_fc, b_fc, out, B, T);
}
